// SAINT_53051436040763
// MI455X (gfx1250) — matrix operand order repaired and checked by reading, not compiled, not run on hardware
//
#include <hip/hip_runtime.h>
#include <hip/hip_fp16.h>

// Problem sizes (fixed by the reference)
#define N_NODES 50000
#define N_EDGES 800000
#define F_H     96     // IN == H == 96
#define F_OUT   64

typedef _Float16 h8  __attribute__((ext_vector_type(8)));
typedef _Float16 h16 __attribute__((ext_vector_type(16)));
typedef float    f8  __attribute__((ext_vector_type(8)));

union AF { h16 v; h8 g[2]; };

// -------- packed-f16 atomic add (GLOBAL_ATOMIC_PK_ADD_F16, no return) --------
__device__ __forceinline__ void atomic_pk_add_f16(uint32_t* addr, uint32_t val) {
    asm volatile("global_atomic_pk_add_f16 %0, %1, off" :: "v"(addr), "v"(val) : "memory");
}

// ---------------------------------------------------------------------------
// K0a: per-row alpha = mean(|W|) for the 4 binarized weight matrices
// ---------------------------------------------------------------------------
__global__ void k_alpha(const float* __restrict__ w0, const float* __restrict__ w1,
                        const float* __restrict__ w2, const float* __restrict__ w3,
                        float* __restrict__ alpha) {
    const float* W = (blockIdx.x == 0) ? w0 : (blockIdx.x == 1) ? w1
                   : (blockIdx.x == 2) ? w2 : w3;
    int n = threadIdx.x;   // 96 threads
    float s = 0.f;
    for (int k = 0; k < F_H; ++k) s += fabsf(W[n * F_H + k]);
    alpha[blockIdx.x * F_H + n] = s * (1.0f / 96.0f);
}

// ---------------------------------------------------------------------------
// K0b: sign(W) -> pre-swizzled f16 WMMA B-fragments (±1).
// B frag layout (16x16x32 f16): lane L -> col n = L%16, k = (h/8)*16 + (L/16)*8 + (h%8).
// Stored so each lane reads one contiguous 32B h16.
// ---------------------------------------------------------------------------
__global__ void k_wfrag(const float* __restrict__ w0, const float* __restrict__ w1,
                        const float* __restrict__ w2, const float* __restrict__ w3,
                        _Float16* __restrict__ frags) {
    int t = blockIdx.x * blockDim.x + threadIdx.x;     // 4*6*3*32*16 = 36864
    if (t >= 36864) return;
    int h    = t & 15;
    int lane = (t >> 4) & 31;
    int kc   = (t >> 9) % 3;
    int nt   = (t / 1536) % 6;
    int w    = t / 9216;
    const float* W = (w == 0) ? w0 : (w == 1) ? w1 : (w == 2) ? w2 : w3;
    int n = nt * 16 + (lane & 15);
    int k = kc * 32 + ((h >> 3) << 4) + ((lane >> 4) << 3) + (h & 7);
    float v = W[n * F_H + k];                           // W is [96 x 96], row-major
    frags[(size_t)w * 9216 + ((size_t)(nt * 3 + kc) * 32 + lane) * 16 + h] =
        (_Float16)((v > 0.f) ? 1.f : (v < 0.f ? -1.f : 0.f));
}

// ---------------------------------------------------------------------------
// K0c: w_lin (64 x 192) -> hi/lo f16 B-fragments (error-compensated split)
// ---------------------------------------------------------------------------
__global__ void k_linfrag(const float* __restrict__ wlin,
                          _Float16* __restrict__ hi, _Float16* __restrict__ lo) {
    int t = blockIdx.x * blockDim.x + threadIdx.x;     // 4*6*32*16 = 12288
    if (t >= 12288) return;
    int h    = t & 15;
    int lane = (t >> 4) & 31;
    int kc   = (t >> 9) % 6;
    int nt   = t / 3072;
    int n = nt * 16 + (lane & 15);
    int k = kc * 32 + ((h >> 3) << 4) + ((lane >> 4) << 3) + (h & 7);
    float v = wlin[n * 192 + k];
    _Float16 vh = (_Float16)v;
    size_t idx = ((size_t)(nt * 6 + kc) * 32 + lane) * 16 + h;
    hi[idx] = vh;
    lo[idx] = (_Float16)(v - (float)vh);
}

// ---------------------------------------------------------------------------
// K1: BinActive input: xb = sign(x0 - rowmean)  (sd+eps > 0, so sd is moot)
// One wave per node; each lane owns 3 of the 96 features.
// ---------------------------------------------------------------------------
__global__ void k_bin(const float* __restrict__ x0, _Float16* __restrict__ xb) {
    int wave = (int)((blockIdx.x * (size_t)blockDim.x + threadIdx.x) >> 5);
    int lane = threadIdx.x & 31;
    if (wave >= N_NODES) return;
    const float* row = x0 + (size_t)wave * F_H + lane * 3;
    float a0 = row[0], a1 = row[1], a2 = row[2];
    float s = a0 + a1 + a2;
    #pragma unroll
    for (int m = 16; m >= 1; m >>= 1) s += __shfl_xor(s, m, 32);
    float mu = s * (1.0f / 96.0f);
    _Float16* o = xb + (size_t)wave * F_H + lane * 3;
    float d0 = a0 - mu, d1 = a1 - mu, d2 = a2 - mu;
    o[0] = (_Float16)((d0 > 0.f) ? 1.f : (d0 < 0.f ? -1.f : 0.f));
    o[1] = (_Float16)((d1 > 0.f) ? 1.f : (d1 < 0.f ? -1.f : 0.f));
    o[2] = (_Float16)((d2 > 0.f) ? 1.f : (d2 < 0.f ? -1.f : 0.f));
}

// ---------------------------------------------------------------------------
// K2: edge gather + scatter-add:  agg[dst] += x[src]   (packed f16 atomics)
// 16 threads per edge, 6 features (3 packed atomics) each. Exact: integer
// sums of ±1 in f16 are exact up to |2048|.
// ---------------------------------------------------------------------------
__global__ void k_scatter(const _Float16* __restrict__ x, const int* __restrict__ src,
                          const int* __restrict__ dst, _Float16* __restrict__ agg) {
    long long t = blockIdx.x * (long long)blockDim.x + threadIdx.x;
    int e = (int)(t >> 4);
    if (e >= N_EDGES) return;
    int part = (int)t & 15;
    const uint32_t* xr = (const uint32_t*)(x + (size_t)src[e] * F_H) + part * 3;
    uint32_t* ar = (uint32_t*)(agg + (size_t)dst[e] * F_H) + part * 3;
    #pragma unroll
    for (int i = 0; i < 3; ++i) atomic_pk_add_f16(ar + i, xr[i]);
}

// ---------------------------------------------------------------------------
// K3: one BiGraphConv layer via WMMA:
//   out = alpha_rel*(agg @ signW_rel^T) + alpha_root*(xin @ signW_root^T) + b
// One wave computes a 16-node x 96-col tile: 3 K-chunks x 6 N-tiles x 2 chains
// = 36 v_wmma_f32_16x16x32_f16. Optionally writes sign(out) for next layer.
// A frag layout: lane L -> row m = L%16, k = (h/8)*16 + (L/16)*8 + (h%8).
// ---------------------------------------------------------------------------
__global__ void k_layer(const _Float16* __restrict__ agg, const _Float16* __restrict__ xin,
                        const h16* __restrict__ frel, const h16* __restrict__ froot,
                        const float* __restrict__ a_rel, const float* __restrict__ a_root,
                        const float* __restrict__ bias, float* __restrict__ xout,
                        _Float16* __restrict__ xsign, int write_sign) {
    int wave = (int)((blockIdx.x * (size_t)blockDim.x + threadIdx.x) >> 5);
    int lane = threadIdx.x & 31;
    if (wave >= N_NODES / 16) return;                   // wave-uniform exit
    int m   = lane & 15;
    int khi = (lane >> 4) << 3;
    size_t rowA = (size_t)(wave * 16 + m) * F_H;

    f8 acc_rel[6] = {}; f8 acc_root[6] = {};
    #pragma unroll
    for (int kc = 0; kc < 3; ++kc) {
        AF aA, aX;
        int c0 = kc * 32 + khi;
        aA.g[0] = *(const h8*)(agg + rowA + c0);
        aA.g[1] = *(const h8*)(agg + rowA + c0 + 16);
        aX.g[0] = *(const h8*)(xin + rowA + c0);
        aX.g[1] = *(const h8*)(xin + rowA + c0 + 16);
        #pragma unroll
        for (int nt = 0; nt < 6; ++nt) {
            h16 br = frel [(nt * 3 + kc) * 32 + lane];
            h16 bo = froot[(nt * 3 + kc) * 32 + lane];
            acc_rel[nt]  = __builtin_amdgcn_wmma_f32_16x16x32_f16(
                false, aA.v, false, br, (short)0, acc_rel[nt],  false, false);
            acc_root[nt] = __builtin_amdgcn_wmma_f32_16x16x32_f16(
                false, aX.v, false, bo, (short)0, acc_root[nt], false, false);
        }
    }
    int mbase = (lane >> 4) << 3;                       // C/D: VGPR r -> M = r (+8 hi half)
    int ncol  = lane & 15;
    #pragma unroll
    for (int nt = 0; nt < 6; ++nt) {
        int n = nt * 16 + ncol;
        float ar = a_rel[n], ao = a_root[n], b = bias[n];
        #pragma unroll
        for (int r = 0; r < 8; ++r) {
            size_t node = (size_t)wave * 16 + mbase + r;
            float v = ar * acc_rel[nt][r] + ao * acc_root[nt][r] + b;
            xout[node * F_H + n] = v;
            if (write_sign)
                xsign[node * F_H + n] =
                    (_Float16)((v > 0.f) ? 1.f : (v < 0.f ? -1.f : 0.f));
        }
    }
}

// ---------------------------------------------------------------------------
// K4: head:  logits = [x1|x2] @ w_lin^T + b_lin ; out = log_softmax(logits)
// hi/lo f16 split on both A and B (drop lo*lo): 3 WMMAs per (ntile,kchunk).
// One wave: 16 nodes x 64 outputs = 4 N-tiles x 6 K-chunks x 3 = 72 WMMAs.
// ---------------------------------------------------------------------------
__global__ void k_final(const float* __restrict__ x1, const float* __restrict__ x2,
                        const h16* __restrict__ bhi, const h16* __restrict__ blo,
                        const float* __restrict__ b_lin, float* __restrict__ out) {
    __shared__ float lds[8][16][F_OUT + 1];
    int wid  = threadIdx.x >> 5;
    int wave = (int)((blockIdx.x * (size_t)blockDim.x + threadIdx.x) >> 5);
    int lane = threadIdx.x & 31;
    bool active = wave < (N_NODES / 16);
    int tile = active ? wave : 0;                       // clamp: loads stay in-bounds
    int m   = lane & 15;
    int khi = (lane >> 4) << 3;
    size_t rowA = (size_t)(tile * 16 + m) * F_H;

    f8 acc[4] = {};
    #pragma unroll
    for (int kc = 0; kc < 6; ++kc) {
        const float* srcb = (kc < 3) ? x1 : x2;         // concat [x1|x2], K chunks never straddle
        int c0 = (kc % 3) * 32 + khi;
        AF ahi, alo;
        #pragma unroll
        for (int g = 0; g < 2; ++g) {
            #pragma unroll
            for (int h = 0; h < 8; ++h) {
                float f = srcb[rowA + c0 + g * 16 + h];
                _Float16 fh = (_Float16)f;
                ahi.g[g][h] = fh;
                alo.g[g][h] = (_Float16)(f - (float)fh);
            }
        }
        #pragma unroll
        for (int nt = 0; nt < 4; ++nt) {
            h16 bh = bhi[(nt * 6 + kc) * 32 + lane];
            h16 bl = blo[(nt * 6 + kc) * 32 + lane];
            acc[nt] = __builtin_amdgcn_wmma_f32_16x16x32_f16(
                false, ahi.v, false, bh, (short)0, acc[nt], false, false);
            acc[nt] = __builtin_amdgcn_wmma_f32_16x16x32_f16(
                false, ahi.v, false, bl, (short)0, acc[nt], false, false);
            acc[nt] = __builtin_amdgcn_wmma_f32_16x16x32_f16(
                false, alo.v, false, bh, (short)0, acc[nt], false, false);
        }
    }
    int mbase = (lane >> 4) << 3;
    int ncol  = lane & 15;
    #pragma unroll
    for (int nt = 0; nt < 4; ++nt) {
        float b = b_lin[nt * 16 + ncol];
        #pragma unroll
        for (int r = 0; r < 8; ++r)
            lds[wid][mbase + r][nt * 16 + ncol] = acc[nt][r] + b;
    }
    __syncthreads();

    // log_softmax: lane pair (L, L^16) shares row L%16, 32 cols each
    int row   = lane & 15;
    int cbase = (lane >> 4) * 32;
    float mx = -3.4e38f;
    #pragma unroll
    for (int c = 0; c < 32; ++c) mx = fmaxf(mx, lds[wid][row][cbase + c]);
    mx = fmaxf(mx, __shfl_xor(mx, 16, 32));
    float se = 0.f;
    #pragma unroll
    for (int c = 0; c < 32; ++c) se += expf(lds[wid][row][cbase + c] - mx);
    se += __shfl_xor(se, 16, 32);
    float lse = logf(se) + mx;
    if (active) {
        size_t node = (size_t)tile * 16 + row;
        #pragma unroll
        for (int c = 0; c < 32; ++c)
            out[node * F_OUT + cbase + c] = lds[wid][row][cbase + c] - lse;
    }
}

// ---------------------------------------------------------------------------
extern "C" void kernel_launch(void* const* d_in, const int* in_sizes, int n_in,
                              void* d_out, int out_size, void* d_ws, size_t ws_size,
                              hipStream_t stream) {
    (void)in_sizes; (void)n_in; (void)out_size; (void)ws_size;
    const float* x0      = (const float*)d_in[0];
    const int*   ei      = (const int*)  d_in[1];        // [2, E]: src row then dst row
    const float* w_rel1  = (const float*)d_in[2];
    const float* w_root1 = (const float*)d_in[3];
    const float* b1      = (const float*)d_in[4];
    const float* w_rel2  = (const float*)d_in[5];
    const float* w_root2 = (const float*)d_in[6];
    const float* b2      = (const float*)d_in[7];
    const float* w_lin   = (const float*)d_in[8];
    const float* b_lin   = (const float*)d_in[9];
    float* out = (float*)d_out;

    char* ws = (char*)d_ws;
    size_t off = 0;
    auto alloc = [&](size_t bytes) -> void* {
        off = (off + 255) & ~(size_t)255;
        void* p = ws + off;
        off += bytes;
        return p;
    };
    const size_t NH = (size_t)N_NODES * F_H;
    _Float16* xb    = (_Float16*)alloc(NH * 2);
    _Float16* agg1  = (_Float16*)alloc(NH * 2);
    _Float16* agg2  = (_Float16*)alloc(NH * 2);
    _Float16* xs1   = (_Float16*)alloc(NH * 2);
    float*    x1    = (float*)   alloc(NH * 4);
    float*    x2    = (float*)   alloc(NH * 4);
    float*    alpha = (float*)   alloc(4 * F_H * 4);     // rel1, root1, rel2, root2
    _Float16* wfr   = (_Float16*)alloc(4 * 9216 * 2);    // 4 swizzled sign-weight frag sets
    _Float16* linhi = (_Float16*)alloc(12288 * 2);
    _Float16* linlo = (_Float16*)alloc(12288 * 2);

    const h16* frel1  = (const h16*)(wfr);
    const h16* froot1 = (const h16*)(wfr + 9216);
    const h16* frel2  = (const h16*)(wfr + 2 * 9216);
    const h16* froot2 = (const h16*)(wfr + 3 * 9216);

    hipMemsetAsync(agg1, 0, NH * 2, stream);
    hipMemsetAsync(agg2, 0, NH * 2, stream);

    k_alpha  <<<4,   96,  0, stream>>>(w_rel1, w_root1, w_rel2, w_root2, alpha);
    k_wfrag  <<<144, 256, 0, stream>>>(w_rel1, w_root1, w_rel2, w_root2, wfr);
    k_linfrag<<<48,  256, 0, stream>>>(w_lin, linhi, linlo);
    k_bin    <<<N_NODES / 8, 256, 0, stream>>>(x0, xb);

    const int* src = ei;
    const int* dst = ei + N_EDGES;
    const int tiles_blocks = (N_NODES / 16 + 7) / 8;     // 8 waves / block, 1 tile / wave

    k_scatter<<<(N_EDGES * 16) / 256, 256, 0, stream>>>(xb, src, dst, agg1);
    k_layer  <<<tiles_blocks, 256, 0, stream>>>(agg1, xb, frel1, froot1,
                                                alpha, alpha + F_H, b1, x1, xs1, 1);
    k_scatter<<<(N_EDGES * 16) / 256, 256, 0, stream>>>(xs1, src, dst, agg2);
    k_layer  <<<tiles_blocks, 256, 0, stream>>>(agg2, xs1, frel2, froot2,
                                                alpha + 2 * F_H, alpha + 3 * F_H, b2,
                                                x2, xs1, 0);
    k_final  <<<tiles_blocks, 256, 0, stream>>>(x1, x2, (const h16*)linhi,
                                                (const h16*)linlo, b_lin, out);
}
